// Atten_SeqSeq_56607668961684
// MI455X (gfx1250) — matrix operand order repaired and checked by reading, not compiled, not run on hardware
//
#include <hip/hip_runtime.h>
#include <math.h>

typedef __attribute__((ext_vector_type(16))) _Float16 v16h;
typedef __attribute__((ext_vector_type(8)))  float    v8f;

#define HDIM 128
#define TSEQ 1024

__device__ __forceinline__ _Float16 f2h(float x) { return (_Float16)x; }

// ---------------- WMMA fragment loaders (f32 memory -> f16 fragments) -------
// A 16x32 f16 layout (ISA 7.12.2): lane = h*16 + m ; VGPR v holds
// K = (v>>2)*16 + h*8 + (v&3)*2  and +1   (consecutive pair -> float2 load)
template<bool GUARD>
__device__ __forceinline__ v16h load_a_frag(const float* __restrict__ Arow,
                                            int kbase, int h, int K) {
  v16h a;
#pragma unroll
  for (int v = 0; v < 8; ++v) {
    int k = kbase + ((v >> 2) << 4) + (h << 3) + ((v & 3) << 1);
    float x0 = 0.f, x1 = 0.f;
    if (!GUARD) {
      float2 t = *(const float2*)(Arow + k); x0 = t.x; x1 = t.y;
    } else {
      if (k + 1 < K)      { float2 t = *(const float2*)(Arow + k); x0 = t.x; x1 = t.y; }
      else if (k < K)     { x0 = Arow[k]; }
    }
    a[2 * v] = f2h(x0); a[2 * v + 1] = f2h(x1);
  }
  return a;
}

// B 32x16: lane = h*16 + n ; VGPR v holds K = (v>>2)*16 + h*8 + (v&3)*2, +1  for column n.
// TRANSB case: B is W[N][K] row-major -> k consecutive -> float2 loads.
template<bool GUARD>
__device__ __forceinline__ v16h load_bT_frag(const float* __restrict__ Wrow,
                                             int kbase, int h, int K) {
  v16h b;
  int kb = kbase + (h << 3);
#pragma unroll
  for (int v = 0; v < 8; ++v) {
    int k = kb + ((v >> 2) << 4) + ((v & 3) << 1);
    float x0 = 0.f, x1 = 0.f;
    if (!GUARD) {
      float2 t = *(const float2*)(Wrow + k); x0 = t.x; x1 = t.y;
    } else {
      if (k + 1 < K)      { float2 t = *(const float2*)(Wrow + k); x0 = t.x; x1 = t.y; }
      else if (k < K)     { x0 = Wrow[k]; }
    }
    b[2 * v] = f2h(x0); b[2 * v + 1] = f2h(x1);
  }
  return b;
}

// Non-transposed B[K][N] row-major: element (k,n) at Bcol[k*ldb] (strided).
template<bool GUARD>
__device__ __forceinline__ v16h load_bN_frag(const float* __restrict__ Bcol,
                                             int ldb, int kbase, int h, int K) {
  v16h b;
  int kb = kbase + (h << 3);
#pragma unroll
  for (int v = 0; v < 8; ++v) {
    int k = kb + ((v >> 2) << 4) + ((v & 3) << 1);
    float x0 = (!GUARD || k     < K) ? Bcol[(size_t)k       * ldb] : 0.f;
    float x1 = (!GUARD || k + 1 < K) ? Bcol[(size_t)(k + 1) * ldb] : 0.f;
    b[2 * v] = f2h(x0); b[2 * v + 1] = f2h(x1);
  }
  return b;
}

// C = A[M,K] x (TRANSB ? B[N,K]^T : B[K,N]) + bias,  f32 in/out, f16 WMMA core.
// block = 128 threads = 4 waves; each wave owns two 16x16 N-tiles of one
// 16-row M strip.  M % 16 == 0 and N % 16 == 0 for every call here.
template<bool TRANSB>
__global__ void __launch_bounds__(128) gemm_f16_wmma(
    const float* __restrict__ A, const float* __restrict__ B,
    const float* __restrict__ bias, float* __restrict__ C,
    int M, int N, int K, int ldb) {
  const int lane  = threadIdx.x & 31;
  const int wave  = threadIdx.x >> 5;
  const int h     = lane >> 4;       // lane half (ISA K-half select)
  const int nl    = lane & 15;
  const int ntile0 = (blockIdx.y * 4 + wave) * 2;
  if (ntile0 * 16 >= N) return;                 // wave-uniform: EXEC stays full
  const bool has1 = (ntile0 + 1) * 16 < N;
  const int  m    = blockIdx.x * 16;

  const float* Arow = A + (size_t)(m + nl) * K;
  const int n0 = ntile0 * 16 + nl;
  const int n1 = n0 + 16;
  const float* B0 = TRANSB ? (B + (size_t)n0 * ldb) : (B + n0);
  const float* B1 = TRANSB ? (B + (size_t)n1 * ldb) : (B + n1);

  v8f acc0 = {}; v8f acc1 = {};
  const int kfull = K & ~31;
  for (int kb = 0; kb < kfull; kb += 32) {
    if (kb + 1024 < K) __builtin_prefetch(Arow + kb + 1024, 0, 1);
    v16h a  = load_a_frag<false>(Arow, kb, h, K);
    v16h b0 = TRANSB ? load_bT_frag<false>(B0, kb, h, K)
                     : load_bN_frag<false>(B0, ldb, kb, h, K);
    acc0 = __builtin_amdgcn_wmma_f32_16x16x32_f16(false, a, false, b0,
                                                  (short)0, acc0, false, false);
    if (has1) {
      v16h b1 = TRANSB ? load_bT_frag<false>(B1, kb, h, K)
                       : load_bN_frag<false>(B1, ldb, kb, h, K);
      acc1 = __builtin_amdgcn_wmma_f32_16x16x32_f16(false, a, false, b1,
                                                    (short)0, acc1, false, false);
    }
  }
  if (kfull < K) {                               // K tail (10000 = 312*32 + 16)
    v16h a  = load_a_frag<true>(Arow, kfull, h, K);
    v16h b0 = TRANSB ? load_bT_frag<true>(B0, kfull, h, K)
                     : load_bN_frag<true>(B0, ldb, kfull, h, K);
    acc0 = __builtin_amdgcn_wmma_f32_16x16x32_f16(false, a, false, b0,
                                                  (short)0, acc0, false, false);
    if (has1) {
      v16h b1 = TRANSB ? load_bT_frag<true>(B1, kfull, h, K)
                       : load_bN_frag<true>(B1, ldb, kfull, h, K);
      acc1 = __builtin_amdgcn_wmma_f32_16x16x32_f16(false, a, false, b1,
                                                    (short)0, acc1, false, false);
    }
  }

  const float bv0 = bias ? bias[n0] : 0.f;
  const float bv1 = (bias && has1) ? bias[n1] : 0.f;
#pragma unroll
  for (int r = 0; r < 8; ++r) {                  // C layout: row = m + h*8 + r
    int row = m + (h << 3) + r;
    C[(size_t)row * N + n0] = acc0[r] + bv0;
    if (has1) C[(size_t)row * N + n1] = acc1[r] + bv1;
  }
}

// -------- sequential tanh-RNN (encoder then decoder), one 128-thread block --
// Thread i keeps W_hh row i in 128 VGPRs; h broadcast through LDS.
__global__ void __launch_bounds__(128, 1) rnn_both_kernel(
    const float* __restrict__ enc_xp, const float* __restrict__ enc_Whh,
    const float* __restrict__ enc_bhh,
    const float* __restrict__ dec_xp, const float* __restrict__ dec_Whh,
    const float* __restrict__ dec_bhh,
    const float* __restrict__ h0,
    float* __restrict__ enc_out, float* __restrict__ dec_out) {
  __shared__ float hs[HDIM];
  const int i = threadIdx.x;
  float w[HDIM];
#pragma unroll
  for (int j = 0; j < HDIM; ++j) w[j] = enc_Whh[i * HDIM + j];
  float bb = enc_bhh[i];
  hs[i] = h0[i];
  __syncthreads();
  for (int t = 0; t < TSEQ; ++t) {
    float acc = enc_xp[t * HDIM + i] + bb;
#pragma unroll
    for (int j = 0; j < HDIM; ++j) acc = fmaf(w[j], hs[j], acc);
    float hnew = tanhf(acc);
    __syncthreads();
    hs[i] = hnew;
    enc_out[t * HDIM + i] = hnew;
    __syncthreads();
  }
#pragma unroll
  for (int j = 0; j < HDIM; ++j) w[j] = dec_Whh[i * HDIM + j];
  bb = dec_bhh[i];
  for (int t = 0; t < TSEQ; ++t) {
    float acc = dec_xp[t * HDIM + i] + bb;
#pragma unroll
    for (int j = 0; j < HDIM; ++j) acc = fmaf(w[j], hs[j], acc);
    float hnew = tanhf(acc);
    __syncthreads();
    hs[i] = hnew;
    dec_out[t * HDIM + i] = hnew;
    __syncthreads();
  }
}

// ---------------- row softmax over [rows, cols] -----------------------------
__global__ void __launch_bounds__(256) softmax_kernel(
    const float* __restrict__ S, float* __restrict__ P, int cols) {
  __shared__ float buf[256];
  const int row = blockIdx.x;
  const float* s = S + (size_t)row * cols;
  float*       p = P + (size_t)row * cols;
  float m = -1e30f;
  for (int c = threadIdx.x; c < cols; c += 256) m = fmaxf(m, s[c]);
  buf[threadIdx.x] = m; __syncthreads();
  for (int off = 128; off > 0; off >>= 1) {
    if (threadIdx.x < off) buf[threadIdx.x] = fmaxf(buf[threadIdx.x], buf[threadIdx.x + off]);
    __syncthreads();
  }
  m = buf[0]; __syncthreads();
  float sum = 0.f;
  for (int c = threadIdx.x; c < cols; c += 256) {
    float e = __expf(s[c] - m); p[c] = e; sum += e;
  }
  buf[threadIdx.x] = sum; __syncthreads();
  for (int off = 128; off > 0; off >>= 1) {
    if (threadIdx.x < off) buf[threadIdx.x] += buf[threadIdx.x + off];
    __syncthreads();
  }
  const float inv = 1.f / buf[0];
  for (int c = threadIdx.x; c < cols; c += 256) p[c] *= inv;
}

// ---------------- concat [dec_out | context] -> [1024, 256] -----------------
__global__ void __launch_bounds__(256) concat_kernel(
    const float* __restrict__ a, const float* __restrict__ b,
    float* __restrict__ out) {
  int idx = blockIdx.x * 256 + threadIdx.x;     // over 1024*256
  int t = idx >> 8, c = idx & 255;
  out[idx] = (c < HDIM) ? a[t * HDIM + c] : b[t * HDIM + (c - HDIM)];
}

extern "C" void kernel_launch(void* const* d_in, const int* in_sizes, int n_in,
                              void* d_out, int out_size, void* d_ws, size_t ws_size,
                              hipStream_t stream) {
  (void)in_sizes; (void)n_in; (void)out_size; (void)ws_size;
  const float* enc_in  = (const float*)d_in[0];
  const float* hidden  = (const float*)d_in[1];
  const float* dec_in  = (const float*)d_in[2];
  const float* enc_Wih = (const float*)d_in[3];
  const float* enc_Whh = (const float*)d_in[4];
  const float* enc_bih = (const float*)d_in[5];
  const float* enc_bhh = (const float*)d_in[6];
  const float* dec_Wih = (const float*)d_in[7];
  const float* dec_Whh = (const float*)d_in[8];
  const float* dec_bih = (const float*)d_in[9];
  const float* dec_bhh = (const float*)d_in[10];
  const float* att_W   = (const float*)d_in[11];
  const float* att_b   = (const float*)d_in[12];
  const float* fc_W    = (const float*)d_in[13];
  const float* fc_b    = (const float*)d_in[14];

  float* ws      = (float*)d_ws;
  float* enc_xp  = ws;  ws += 1024 * 128;
  float* dec_xp  = ws;  ws += 1024 * 128;
  float* enc_out = ws;  ws += 1024 * 128;
  float* dec_out = ws;  ws += 1024 * 128;
  float* keys    = ws;  ws += 1024 * 128;
  float* scores  = ws;  ws += 1024 * 1024;
  float* context = ws;  ws += 1024 * 128;
  float* catbuf  = ws;  ws += 1024 * 256;

  float* logits = (float*)d_out;
  float* attn   = logits + (size_t)1024 * 10000;

  dim3 blk(128);
  auto grid = [](int M, int N) { return dim3(M / 16, (((N + 15) / 16) + 7) / 8); };

  // input projections (memory-bound: stream 2x41MB through WMMA)
  gemm_f16_wmma<true ><<<grid(1024, 128), blk, 0, stream>>>(enc_in, enc_Wih, enc_bih, enc_xp, 1024, 128, 10000, 10000);
  gemm_f16_wmma<true ><<<grid(1024, 128), blk, 0, stream>>>(dec_in, dec_Wih, dec_bih, dec_xp, 1024, 128, 10000, 10000);
  // sequential RNN (latency-bound)
  rnn_both_kernel<<<1, 128, 0, stream>>>(enc_xp, enc_Whh, enc_bhh,
                                         dec_xp, dec_Whh, dec_bhh,
                                         hidden, enc_out, dec_out);
  // attention
  gemm_f16_wmma<true ><<<grid(1024, 128),  blk, 0, stream>>>(enc_out, att_W, att_b,   keys,   1024, 128,  128,  128);
  gemm_f16_wmma<true ><<<grid(1024, 1024), blk, 0, stream>>>(dec_out, keys,  nullptr, scores, 1024, 1024, 128,  128);
  softmax_kernel<<<1024, 256, 0, stream>>>(scores, attn, 1024);
  gemm_f16_wmma<false><<<grid(1024, 128),  blk, 0, stream>>>(attn, enc_out, nullptr, context, 1024, 128, 1024, 128);
  // output fc
  concat_kernel<<<(1024 * 256) / 256, 256, 0, stream>>>(dec_out, context, catbuf);
  gemm_f16_wmma<true ><<<grid(1024, 10000), blk, 0, stream>>>(catbuf, fc_W, fc_b, logits, 1024, 10000, 256, 256);
}
